// Attention_14826227106248
// MI455X (gfx1250) — matrix operand order repaired and checked by reading, not compiled, not run on hardware
//
#include <hip/hip_runtime.h>

typedef unsigned short u16;
typedef unsigned int   u32;
typedef __attribute__((ext_vector_type(16))) __bf16        v16bf;
typedef __attribute__((ext_vector_type(8)))  float         v8f;
typedef __attribute__((ext_vector_type(4)))  unsigned int  uv4;
typedef __attribute__((ext_vector_type(4)))  float         fv4;
typedef __attribute__((ext_vector_type(4)))  int           i32x4;

#define HEADS 12
#define HD    64
#define NTOK  577
#define NPAD  608            // 19*32 key tiles; q uses 37*16=592 of it
#define BATCH 32
#define DIM   768
#define MTOT  (BATCH*NTOK)   // 18464
#define MBLK  ((MTOT + 127) / 128)   // 145 M-blocks of 128 rows
#define BSTRIDE 40           // LDS B-tile row stride (bf16), kills bank conflicts

// ---- CDNA5 async-to-LDS feature detection (compile-safe fallback) ----
#if defined(__HIP_DEVICE_COMPILE__) && defined(__gfx1250__)
#if __has_builtin(__builtin_amdgcn_global_load_async_to_lds_b128) && \
    __has_builtin(__builtin_amdgcn_s_wait_asynccnt)
#define HAVE_ASYNC_LDS 1
#endif
#endif

// ---------- helpers ----------
__device__ __forceinline__ u16 f2bf(float f) {
  u32 u = __float_as_uint(f);
  return (u16)((u + (((u >> 16) & 1u) + 0x7FFFu)) >> 16);   // RNE
}

union BFrag { v16bf v; u16 u[16]; uv4 q[2]; };
union F8    { v8f v; float f[8]; };

// 16 contiguous bf16 (B-matrix fragment half: K = i + half*16)
__device__ __forceinline__ v16bf ld_frag_contig(const u16* p) {
  BFrag fr;
  fr.q[0] = *(const uv4*)(p);
  fr.q[1] = *(const uv4*)(p + 8);
  return fr.v;
}
// A-matrix fragment: two 8-element chunks (K = {half*16..+8, half*16+8..+8})
__device__ __forceinline__ v16bf ld_fragA(const u16* p0, const u16* p1) {
  BFrag fr;
  fr.q[0] = *(const uv4*)(p0);
  fr.q[1] = *(const uv4*)(p1);
  return fr.v;
}
__device__ __forceinline__ v8f wmma_bf16(v16bf a, v16bf b, v8f c) {
  return __builtin_amdgcn_wmma_f32_16x16x32_bf16(false, a, false, b, (short)0, c, false, false);
}
__device__ __forceinline__ float redmax16(float v) {
  v = fmaxf(v, __shfl_xor(v, 1));
  v = fmaxf(v, __shfl_xor(v, 2));
  v = fmaxf(v, __shfl_xor(v, 4));
  v = fmaxf(v, __shfl_xor(v, 8));
  return v;
}
__device__ __forceinline__ float redsum16(float v) {
  v += __shfl_xor(v, 1);
  v += __shfl_xor(v, 2);
  v += __shfl_xor(v, 4);
  v += __shfl_xor(v, 8);
  return v;
}

// Copy 32B (16 bf16) global -> LDS: async-to-LDS on CDNA5, sync fallback otherwise.
__device__ __forceinline__ void copy32_g2l(const u16* __restrict__ gp, u16* lp) {
#if defined(HAVE_ASYNC_LDS)
  __builtin_amdgcn_global_load_async_to_lds_b128(
      (__attribute__((address_space(1))) i32x4*)gp,
      (__attribute__((address_space(3))) i32x4*)lp, 0, 0);
  __builtin_amdgcn_global_load_async_to_lds_b128(
      (__attribute__((address_space(1))) i32x4*)(gp + 8),
      (__attribute__((address_space(3))) i32x4*)(lp + 8), 0, 0);
#else
  *(uv4*)lp = *(const uv4*)gp;
  *(uv4*)(lp + 8) = *(const uv4*)(gp + 8);
#endif
}
__device__ __forceinline__ void wait_async_all() {
#if defined(HAVE_ASYNC_LDS)
  __builtin_amdgcn_s_wait_asynccnt(0);
#endif
}

// Stage one 128x32 bf16 B-tile (cols nbase..+127, k kk..+31) into LDS (stride BSTRIDE).
__device__ __forceinline__ void stage_b(const u16* __restrict__ wb, u16* tile,
                                        int nbase, int kk, int tid) {
  const int n = tid >> 1, hh = tid & 1;      // 256 threads: 128 rows x 2 half-rows
  const u16* gp = wb + (size_t)(nbase + n) * DIM + kk + hh * 16;
  u16* lp = tile + n * BSTRIDE + hh * 16;
  copy32_g2l(gp, lp);
}

// ---------- precompute kernels ----------
__global__ void cvt_bf16(const float* __restrict__ s, u16* __restrict__ d, int n) {
  int i = blockIdx.x * blockDim.x + threadIdx.x;
  if (i < n) d[i] = f2bf(s[i]);
}

__global__ void bias_gather(const float* __restrict__ table, const int* __restrict__ rel,
                            float* __restrict__ bias) {
  int i = blockIdx.x * blockDim.x + threadIdx.x;
  const int NN = NTOK * NTOK;
  if (i >= HEADS * NN) return;
  int h = i / NN, qk = i - h * NN;
  bias[(size_t)h * NN + qk] = table[rel[qk] * HEADS + h];
}

// ---------- QKV GEMM: [18464,768] x [2304,768]^T, LDS-staged B, scatter to Q/K/V^T ----------
__global__ void __launch_bounds__(256) qkv_gemm(const u16* __restrict__ xb, const u16* __restrict__ wb,
                                                u16* __restrict__ qb, u16* __restrict__ kb,
                                                u16* __restrict__ vtb) {
  __shared__ u16 tile[2][128 * BSTRIDE];
  const int tid = threadIdx.x;
  const int bid = blockIdx.x;
  const int mblk = bid / 18, nblk = bid - mblk * 18;
  const int nbase = nblk * 128;
  const int wv = tid >> 5, lane = tid & 31, half = lane >> 4, lq = lane & 15;
  const int mrow0 = mblk * 128 + wv * 16;
  int arowi = mrow0 + lq; if (arowi >= MTOT) arowi = MTOT - 1;   // clamp: keeps EXEC full
  const u16* arow = xb + (size_t)arowi * DIM;

  F8 z;
  #pragma unroll
  for (int i = 0; i < 8; ++i) z.f[i] = 0.0f;
  v8f acc[8];
  #pragma unroll
  for (int t = 0; t < 8; ++t) acc[t] = z.v;

  stage_b(wb, tile[0], nbase, 0, tid);
  for (int ki = 0; ki < 24; ++ki) {
    wait_async_all();
    __syncthreads();
    if (ki < 23) stage_b(wb, tile[(ki + 1) & 1], nbase, (ki + 1) * 32, tid);
    const u16* bt = tile[ki & 1];
    const int kk = ki * 32;
    v16bf a = ld_fragA(arow + kk + half * 16, arow + kk + half * 16 + 8);
    #pragma unroll
    for (int t = 0; t < 8; ++t) {
      v16bf b = ld_frag_contig(bt + (t * 16 + lq) * BSTRIDE + half * 16);
      acc[t] = wmma_bf16(a, b, acc[t]);
    }
  }

  #pragma unroll
  for (int r = 0; r < 8; ++r) {
    const int gm = mrow0 + r + 8 * half;
    if (gm >= MTOT) continue;
    const int bb = gm / NTOK, nn = gm - bb * NTOK;
    #pragma unroll
    for (int t = 0; t < 8; ++t) {
      F8 c; c.v = acc[t];
      const int n = nbase + t * 16 + lq;
      const int which = n / DIM, rem = n - which * DIM;
      const int h = rem >> 6, d = rem & 63;
      const int bh = bb * HEADS + h;
      const float val = c.f[r];
      if (which == 0)      qb[((size_t)bh * NPAD + nn) * 64 + d] = f2bf(val * 0.125f);
      else if (which == 1) kb[((size_t)bh * NPAD + nn) * 64 + d] = f2bf(val);
      else                 vtb[((size_t)bh * 64 + d) * NPAD + nn] = f2bf(val);
    }
  }
}

// ---------- fused flash attention (16 Q-rows per wave, 32 keys per step) ----------
__global__ void __launch_bounds__(128) flash_attn(const u16* __restrict__ qb, const u16* __restrict__ kb,
                                                  const u16* __restrict__ vtb, const float* __restrict__ bias,
                                                  u16* __restrict__ ob) {
  __shared__ float lds[4 * 16 * 36];
  const int wave = threadIdx.x >> 5;
  const int wid = blockIdx.x * 4 + wave;
  if (wid >= BATCH * HEADS * 37) return;          // wave-uniform
  const int lane = threadIdx.x & 31, half = lane >> 4, lq = lane & 15;
  const int bh = wid / 37, mt = wid - bh * 37;
  const int b = bh / HEADS, h = bh - b * HEADS;
  float* P = lds + wave * 16 * 36;

  const u16* qrow = qb + ((size_t)bh * NPAD + mt * 16 + lq) * 64;
  const v16bf aq0 = ld_fragA(qrow + half * 16,      qrow + half * 16 + 8);
  const v16bf aq1 = ld_fragA(qrow + 32 + half * 16, qrow + 32 + half * 16 + 8);

  float mrun[8], lrun[8];
  #pragma unroll
  for (int r = 0; r < 8; ++r) { mrun[r] = -3.0e38f; lrun[r] = 0.0f; }
  F8 o[4];
  #pragma unroll
  for (int t = 0; t < 4; ++t)
    #pragma unroll
    for (int i = 0; i < 8; ++i) o[t].f[i] = 0.0f;

  F8 z;
  #pragma unroll
  for (int i = 0; i < 8; ++i) z.f[i] = 0.0f;

  const float* biasH = bias + (size_t)h * NTOK * NTOK;
  const u16* vbase = vtb + (size_t)bh * 64 * NPAD;

  for (int it = 0; it < 19; ++it) {
    const int j0 = it * 32;
    const u16* k0 = kb + ((size_t)bh * NPAD + j0 + lq) * 64;
    const u16* k1 = k0 + 16 * 64;
    const v16bf b00 = ld_frag_contig(k0 + half * 16);
    const v16bf b01 = ld_frag_contig(k0 + 32 + half * 16);
    const v16bf b10 = ld_frag_contig(k1 + half * 16);
    const v16bf b11 = ld_frag_contig(k1 + 32 + half * 16);

    F8 s0, s1;
    s0.v = wmma_bf16(aq0, b00, z.v);  s0.v = wmma_bf16(aq1, b01, s0.v);
    s1.v = wmma_bf16(aq0, b10, z.v);  s1.v = wmma_bf16(aq1, b11, s1.v);

    const int c0 = j0 + lq, c1 = c0 + 16;
    const int cc0 = (c0 < NTOK) ? c0 : (NTOK - 1);
    const int cc1 = (c1 < NTOK) ? c1 : (NTOK - 1);

    #pragma unroll
    for (int r = 0; r < 8; ++r) {
      int qi = mt * 16 + r + 8 * half; if (qi >= NTOK) qi = NTOK - 1;
      const float* brow = biasH + (size_t)qi * NTOK;
      float v0 = (c0 < NTOK) ? s0.f[r] + brow[cc0] : -3.0e38f;
      float v1 = (c1 < NTOK) ? s1.f[r] + brow[cc1] : -3.0e38f;
      float rm   = redmax16(fmaxf(v0, v1));
      float mnew = fmaxf(mrun[r], rm);
      float alpha = __expf(mrun[r] - mnew);
      float p0 = __expf(v0 - mnew);
      float p1 = __expf(v1 - mnew);
      float rs = redsum16(p0 + p1);
      lrun[r] = lrun[r] * alpha + rs;
      mrun[r] = mnew;
      #pragma unroll
      for (int t = 0; t < 4; ++t) o[t].f[r] *= alpha;
      const int mloc = r + 8 * half;
      P[mloc * 36 + lq]      = p0;          // D-layout -> LDS tile (16x32, stride 36)
      P[mloc * 36 + 16 + lq] = p1;
    }

    // repack P from LDS into A-fragment bf16 layout (same wave: DS in-order)
    BFrag fr;
    {
      const float* prow = P + lq * 36;
      const fv4 q0 = *(const fv4*)(prow + half * 16);
      const fv4 q1 = *(const fv4*)(prow + half * 16 + 4);
      const fv4 q2 = *(const fv4*)(prow + half * 16 + 8);
      const fv4 q3 = *(const fv4*)(prow + half * 16 + 12);
      #pragma unroll
      for (int i = 0; i < 4; ++i) {
        fr.u[i]      = f2bf(q0[i]);
        fr.u[4 + i]  = f2bf(q1[i]);
        fr.u[8 + i]  = f2bf(q2[i]);
        fr.u[12 + i] = f2bf(q3[i]);
      }
    }
    const v16bf pa = fr.v;

    #pragma unroll
    for (int t = 0; t < 4; ++t) {
      const v16bf bv = ld_frag_contig(vbase + (size_t)(t * 16 + lq) * NPAD + j0 + half * 16);
      o[t].v = wmma_bf16(pa, bv, o[t].v);
    }
  }

  #pragma unroll
  for (int r = 0; r < 8; ++r) {
    const int nn = mt * 16 + r + 8 * half;
    if (nn < NTOK) {
      const float inv = 1.0f / lrun[r];
      const size_t rowoff = ((size_t)(b * NTOK + nn)) * DIM + h * 64;
      #pragma unroll
      for (int t = 0; t < 4; ++t)
        ob[rowoff + t * 16 + lq] = f2bf(o[t].f[r] * inv);
    }
  }
}

// ---------- output projection: [18464,768] x [768,768]^T + bias -> f32, LDS-staged B ----------
__global__ void __launch_bounds__(256) proj_gemm(const u16* __restrict__ ab, const u16* __restrict__ wb,
                                                 const float* __restrict__ pb, float* __restrict__ out) {
  __shared__ u16 tile[2][128 * BSTRIDE];
  const int tid = threadIdx.x;
  const int bid = blockIdx.x;
  const int mblk = bid / 6, nblk = bid - mblk * 6;
  const int nbase = nblk * 128;
  const int wv = tid >> 5, lane = tid & 31, half = lane >> 4, lq = lane & 15;
  const int mrow0 = mblk * 128 + wv * 16;
  int arowi = mrow0 + lq; if (arowi >= MTOT) arowi = MTOT - 1;
  const u16* arow = ab + (size_t)arowi * DIM;

  F8 z;
  #pragma unroll
  for (int i = 0; i < 8; ++i) z.f[i] = 0.0f;
  v8f acc[8];
  #pragma unroll
  for (int t = 0; t < 8; ++t) acc[t] = z.v;

  stage_b(wb, tile[0], nbase, 0, tid);
  for (int ki = 0; ki < 24; ++ki) {
    wait_async_all();
    __syncthreads();
    if (ki < 23) stage_b(wb, tile[(ki + 1) & 1], nbase, (ki + 1) * 32, tid);
    const u16* bt = tile[ki & 1];
    const int kk = ki * 32;
    v16bf a = ld_fragA(arow + kk + half * 16, arow + kk + half * 16 + 8);
    #pragma unroll
    for (int t = 0; t < 8; ++t) {
      v16bf b = ld_frag_contig(bt + (t * 16 + lq) * BSTRIDE + half * 16);
      acc[t] = wmma_bf16(a, b, acc[t]);
    }
  }

  #pragma unroll
  for (int t = 0; t < 8; ++t) {
    F8 c; c.v = acc[t];
    const int n = nbase + t * 16 + lq;
    const float bn = pb[n];
    #pragma unroll
    for (int r = 0; r < 8; ++r) {
      const int gm = mrow0 + r + 8 * half;
      if (gm < MTOT) out[(size_t)gm * DIM + n] = c.f[r] + bn;
    }
  }
}

// ---------- host ----------
extern "C" void kernel_launch(void* const* d_in, const int* in_sizes, int n_in,
                              void* d_out, int out_size, void* d_ws, size_t ws_size,
                              hipStream_t stream) {
  const float* x      = (const float*)d_in[0];
  const float* qkv_w  = (const float*)d_in[1];
  const float* table  = (const float*)d_in[2];
  const int*   rel    = (const int*)d_in[3];
  const float* proj_w = (const float*)d_in[4];
  const float* proj_b = (const float*)d_in[5];
  float* out = (float*)d_out;

  char* ws = (char*)d_ws;
  size_t off = 0;
  auto take = [&](size_t bytes) -> char* {
    char* p = ws + off;
    off = (off + bytes + 255) & ~(size_t)255;
    return p;
  };
  u16* xb     = (u16*)take((size_t)MTOT * DIM * 2);
  u16* wqkvb  = (u16*)take((size_t)3 * DIM * DIM * 2);
  u16* wprojb = (u16*)take((size_t)DIM * DIM * 2);
  u16* qb     = (u16*)take((size_t)BATCH * HEADS * NPAD * 64 * 2);
  u16* kb     = (u16*)take((size_t)BATCH * HEADS * NPAD * 64 * 2);
  u16* vtb    = (u16*)take((size_t)BATCH * HEADS * 64 * NPAD * 2);
  float* bias = (float*)take((size_t)HEADS * NTOK * NTOK * 4);
  u16* aob    = (u16*)take((size_t)MTOT * DIM * 2);

  const int n_x  = MTOT * DIM;
  const int n_wq = 3 * DIM * DIM;
  const int n_wp = DIM * DIM;
  const int n_bi = HEADS * NTOK * NTOK;

  cvt_bf16<<<(n_x + 255) / 256, 256, 0, stream>>>(x, xb, n_x);
  cvt_bf16<<<(n_wq + 255) / 256, 256, 0, stream>>>(qkv_w, wqkvb, n_wq);
  cvt_bf16<<<(n_wp + 255) / 256, 256, 0, stream>>>(proj_w, wprojb, n_wp);
  bias_gather<<<(n_bi + 255) / 256, 256, 0, stream>>>(table, rel, bias);

  qkv_gemm<<<MBLK * 18, 256, 0, stream>>>(xb, wqkvb, qb, kb, vtb);

  const int fa_waves = BATCH * HEADS * 37;  // 14208
  flash_attn<<<(fa_waves + 3) / 4, 128, 0, stream>>>(qb, kb, vtb, bias, aob);

  proj_gemm<<<MBLK * 6, 256, 0, stream>>>(aob, wprojb, proj_b, out);
}
